// TemporalConvTranspose2d_19490561590103
// MI455X (gfx1250) — matrix operand order repaired and checked by reading, not compiled, not run on hardware
//
#include <hip/hip_runtime.h>
#include <hip/hip_bf16.h>

typedef __attribute__((ext_vector_type(16))) _Float16 v16h;
typedef __attribute__((ext_vector_type(8)))  float    v8f;

#define D_IN 32
#define N_BAND_IN 32
#define D_OUT 32
#define N_BAND_OUT 64
#define Bsz 8
#define Tsz 1000
#define T2 1002          // 2 leading zeros for causal pad
#define TTILES 63        // ceil(1000/16)
#define WA_HALVES (4*3*2*32*16)  // kb*kt*mblock*lane*halves = 12288

// ---------------------------------------------------------------------------
// Pack W (f32, [co][ci][kb][kt]) into f16 A-matrix VGPR layout:
// unit index = (((kb*3+kt)*2 + m)*32 + lane)*16 + h
// A 16x32 f16, contiguous k order chosen here (B same): lane<16 -> M=lane, halves {0..15}=K0..15
//                     lane>=16 -> M=lane-16, halves {0..15}=K16..31
// ---------------------------------------------------------------------------
__global__ void pack_w_kernel(const float* __restrict__ W, _Float16* __restrict__ WA) {
    int idx = blockIdx.x * blockDim.x + threadIdx.x;
    if (idx >= WA_HALVES) return;
    int h    = idx & 15;
    int lane = (idx >> 4) & 31;
    int m    = (idx >> 9) & 1;
    int kt   = (idx >> 10) % 3;
    int kb   = (idx >> 10) / 3;
    int hi   = lane >> 4;
    int co   = m * 16 + (lane & 15);
    int ci   = hi * 16 + h;
    WA[idx] = (_Float16)W[((co * D_IN + ci) * 4 + kb) * 3 + kt];
}

// ---------------------------------------------------------------------------
// Pack x (f32, [b][band*32+ci][t]) into f16 [b][band][tp][ci], tp = t+2,
// with 2 leading zero time steps per band. LDS-tiled transpose so both the
// global read (over t) and global write (over ci) are coalesced.
// grid: b(8) * band(32) * tpTile(32), block 256
// ---------------------------------------------------------------------------
__global__ void pack_x_kernel(const float* __restrict__ x, _Float16* __restrict__ xT) {
    __shared__ float tile[32][33];
    int blk    = blockIdx.x;
    int tpTile = blk & 31;
    int band   = (blk >> 5) & 31;
    int b      = blk >> 10;
    int tid    = threadIdx.x;
    int tl     = tid & 31;   // fast index
    int grp    = tid >> 5;   // 0..7
    int tp0    = tpTile * 32;

    #pragma unroll
    for (int i = 0; i < 4; ++i) {
        int ci = grp + i * 8;
        int t  = tp0 + tl - 2;
        float v = 0.0f;
        if (t >= 0 && t < Tsz)
            v = x[(b * (D_IN * N_BAND_IN) + band * D_IN + ci) * Tsz + t];
        tile[ci][tl] = v;
    }
    __syncthreads();
    #pragma unroll
    for (int i = 0; i < 4; ++i) {
        int tlocal = grp + i * 8;
        int tp = tp0 + tlocal;
        if (tp < T2)
            xT[((b * N_BAND_IN + band) * T2 + tp) * D_IN + tl] = (_Float16)tile[tl][tlocal];
    }
}

// ---------------------------------------------------------------------------
// Main kernel: one wave32 per (b, n_out, 16-wide t-tile).
// D tile = 32(co) x 16(t) f32, built from up to 12 v_wmma_f32_16x16x32_f16.
// ---------------------------------------------------------------------------
__global__ void conv_wmma_kernel(const _Float16* __restrict__ WA,
                                 const _Float16* __restrict__ xT,
                                 const float* __restrict__ bias,
                                 float* __restrict__ out) {
    int wave  = blockIdx.x * (blockDim.x >> 5) + (threadIdx.x >> 5);
    int lane  = threadIdx.x & 31;
    int tt    = wave % TTILES;
    int rem   = wave / TTILES;
    int n_out = rem & 63;
    int b     = rem >> 6;
    if (b >= Bsz) return;

    int lo = lane & 15;
    int hi = lane >> 4;
    int t0 = tt * 16;
    int t  = t0 + lo;

    v8f c0 = {};
    v8f c1 = {};

    // Band taps: exactly two candidate (kb, n_in) pairs per n_out (wave-uniform)
    int parity = n_out & 1;
    int kbs[2], nins[2], valid[2];
    kbs[0]  = parity;
    nins[0] = (n_out + parity - 2) >> 1;
    valid[0] = (n_out >= (parity ? 1 : 2));
    kbs[1]  = parity + 2;
    nins[1] = (n_out + parity) >> 1;
    valid[1] = (nins[1] < N_BAND_IN);

    const v16h* WAv = (const v16h*)WA;

    #pragma unroll
    for (int tap = 0; tap < 2; ++tap) {
        if (!valid[tap]) continue;   // uniform across the wave
        int kb = kbs[tap];
        int xbase = (b * N_BAND_IN + nins[tap]) * T2 * D_IN;
        #pragma unroll
        for (int kt = 0; kt < 3; ++kt) {
            int tp = t0 + lo + kt;           // = (t + kt - 2) + 2
            if (tp > T2 - 1) tp = T2 - 1;    // only hits masked (t>=1000) lanes
            v16h bmat = *(const v16h*)(xT + xbase + tp * D_IN + hi * 16);
            int aunit = ((kb * 3 + kt) * 2) * 32 + lane;
            v16h a0 = WAv[aunit];
            v16h a1 = WAv[aunit + 32];       // m=1 block
            c0 = __builtin_amdgcn_wmma_f32_16x16x32_f16(
                     false, a0, false, bmat, (short)0, c0, false, false);
            c1 = __builtin_amdgcn_wmma_f32_16x16x32_f16(
                     false, a1, false, bmat, (short)0, c1, false, false);
        }
    }

    if (t < Tsz) {
        int obase = (b * (D_OUT * N_BAND_OUT) + n_out * D_OUT) * Tsz + t;
        #pragma unroll
        for (int v = 0; v < 8; ++v) {
            int co0 = v + hi * 8;       // M in first 16x16 block
            int co1 = co0 + 16;         // second block
            out[obase + co0 * Tsz] = c0[v] + bias[co0];
            out[obase + co1 * Tsz] = c1[v] + bias[co1];
        }
    }
}

// ---------------------------------------------------------------------------
extern "C" void kernel_launch(void* const* d_in, const int* in_sizes, int n_in,
                              void* d_out, int out_size, void* d_ws, size_t ws_size,
                              hipStream_t stream) {
    const float* x    = (const float*)d_in[0];   // (8, 1024, 1000)
    const float* W    = (const float*)d_in[1];   // (32, 32, 4, 3)
    const float* bias = (const float*)d_in[2];   // (32,)
    float* out = (float*)d_out;                  // (8, 2048, 1000)

    _Float16* WA = (_Float16*)d_ws;              // 12288 halves = 24 KB
    _Float16* xT = WA + WA_HALVES;               // 8*32*1002*32 halves ~ 16.4 MB

    pack_w_kernel<<<(WA_HALVES + 255) / 256, 256, 0, stream>>>(W, WA);
    pack_x_kernel<<<Bsz * N_BAND_IN * 32, 256, 0, stream>>>(x, xT);

    int total_waves = TTILES * N_BAND_OUT * Bsz;   // 32256
    int blocks = total_waves / 8;                  // 8 waves per 256-thread block
    conv_wmma_kernel<<<blocks, 256, 0, stream>>>(WA, xT, bias, out);
}
